// D1Model1_22883585753474
// MI455X (gfx1250) — matrix operand order repaired and checked by reading, not compiled, not run on hardware
//
#include <hip/hip_runtime.h>
#include <hip/hip_bf16.h>

// Problem constants (match reference)
#define NNODES 65536
#define BGRAPH 2048
#define NPGC   32
#define DIM    64
#define NSUB   524288    // S
#define SEDGE  1048576   // SE
#define LLAYERS 4
#define DFFC   512
#define HEADSC 8
#define DHC    8

typedef __attribute__((ext_vector_type(16))) __bf16 v16bf;
typedef __attribute__((ext_vector_type(8)))  float  v8f;

__device__ __forceinline__ unsigned short f2bf(float f) {
  unsigned int u = __float_as_uint(f);
  unsigned int r = (u + 0x7FFFu + ((u >> 16) & 1u)) >> 16;  // round-to-nearest-even
  return (unsigned short)r;
}
__device__ __forceinline__ unsigned int pack2(float a, float b) {
  return (unsigned int)f2bf(a) | ((unsigned int)f2bf(b) << 16);
}

union AFrag { v16bf v; unsigned int u[8]; };

// C[M,Nn] = epilogue( (A [+ A2]) @ W )
// epilogue: +bias -> *scale+shift -> +res -> relu   (or LayerNorm if lng != null)
// Requires: M % 16 == 0, Nn % 64 == 0.  KAL => K % 32 == 0 (wide unguarded loads).
template <bool HAS_A2, bool KAL>
__global__ __launch_bounds__(128)
void gemm_bf16_kernel(const float* __restrict__ A, const float* __restrict__ A2,
                      const float* __restrict__ W, const float* __restrict__ bias,
                      const float* __restrict__ res,
                      const float* __restrict__ scale, const float* __restrict__ shift,
                      const float* __restrict__ lng, const float* __restrict__ lnb,
                      float* __restrict__ C, int M, int K, int Nn, int relu)
{
  // LDS tiles stored as packed bf16 k-pairs, laid out for WMMA fragment reads.
  __shared__ unsigned int sAp[16 * 20];   // [m][kpair], stride 20 (16 used), 16B-aligned rows
  __shared__ unsigned int sWp[16 * 68];   // [kpair][col], stride 68 (64 used), 16B-aligned rows
  __shared__ float sC[16 * 64];
  __shared__ float sMu[16], sRs[16];

  const int tid  = threadIdx.x;
  const int wave = tid >> 5;
  const int lane = tid & 31;
  const int lm   = lane & 15;
  const int lh   = lane >> 4;
  const int rowBase = blockIdx.x * 16;
  const int colBase = blockIdx.y * 64;

  v8f acc = {0.f, 0.f, 0.f, 0.f, 0.f, 0.f, 0.f, 0.f};

  const int ksteps = (K + 31) >> 5;
  for (int kt = 0; kt < ksteps; ++kt) {
    const int k0 = kt << 5;

    // ---- stage A tile (16 rows x 32 k): one float4 per thread ----
    {
      const int r  = tid >> 3;            // 0..15
      const int c4 = (tid & 7) << 2;      // 0,4,...,28
      const size_t gbase = (size_t)(rowBase + r) * K + k0 + c4;
      float e0, e1, e2, e3;
      if (KAL) {
        float4 v = *(const float4*)(A + gbase);
        if (HAS_A2) {
          float4 w = *(const float4*)(A2 + gbase);
          v.x += w.x; v.y += w.y; v.z += w.z; v.w += w.w;
        }
        e0 = v.x; e1 = v.y; e2 = v.z; e3 = v.w;
        if (kt + 1 < ksteps)
          __builtin_prefetch(A + gbase + 32, 0, 1);   // global_prefetch_b8
      } else {
        float t[4];
#pragma unroll
        for (int j = 0; j < 4; ++j) {
          const int k = k0 + c4 + j;
          float v = 0.f;
          if (k < K) {
            v = A[(size_t)(rowBase + r) * K + k];
            if (HAS_A2) v += A2[(size_t)(rowBase + r) * K + k];
          }
          t[j] = v;
        }
        e0 = t[0]; e1 = t[1]; e2 = t[2]; e3 = t[3];
      }
      *(uint2*)&sAp[r * 20 + (c4 >> 1)] = make_uint2(pack2(e0, e1), pack2(e2, e3));
    }

    // ---- stage W tile (32 k x 64 cols) as kpair-packed: 2 x (2 float4 loads + b128 store) ----
#pragma unroll
    for (int seg = 0; seg < 2; ++seg) {
      const int linear = tid + (seg << 7);   // 0..255
      const int rr = linear >> 4;            // kpair row 0..15
      const int cc = (linear & 15) << 2;     // col 0,4,...,60
      const int ka = k0 + (rr << 1);
      float4 r0, r1;
      if (KAL) {
        r0 = *(const float4*)(W + (size_t)ka * Nn + colBase + cc);
        r1 = *(const float4*)(W + (size_t)(ka + 1) * Nn + colBase + cc);
      } else {
        r0 = (ka     < K) ? *(const float4*)(W + (size_t)ka * Nn + colBase + cc)
                          : make_float4(0.f, 0.f, 0.f, 0.f);
        r1 = (ka + 1 < K) ? *(const float4*)(W + (size_t)(ka + 1) * Nn + colBase + cc)
                          : make_float4(0.f, 0.f, 0.f, 0.f);
      }
      *(uint4*)&sWp[rr * 68 + cc] = make_uint4(pack2(r0.x, r1.x), pack2(r0.y, r1.y),
                                               pack2(r0.z, r1.z), pack2(r0.w, r1.w));
    }
    __syncthreads();

    // ---- build fragments (A: two contiguous b128 LDS loads; B: 8 b32 loads) ----
    AFrag fa, fb;
    {
      const uint4 a0 = *(const uint4*)&sAp[lm * 20 + (lh << 2)];        // kpair lh*4+0..3
      const uint4 a1 = *(const uint4*)&sAp[lm * 20 + 8 + (lh << 2)];    // kpair 8+lh*4+0..3
      fa.u[0] = a0.x; fa.u[1] = a0.y; fa.u[2] = a0.z; fa.u[3] = a0.w;
      fa.u[4] = a1.x; fa.u[5] = a1.y; fa.u[6] = a1.z; fa.u[7] = a1.w;
    }
#pragma unroll
    for (int vv = 0; vv < 8; ++vv)
      fb.u[vv] = sWp[(((vv >> 2) << 3) + (lh << 2) + (vv & 3)) * 68 + (wave << 4) + lm];

    acc = __builtin_amdgcn_wmma_f32_16x16x32_bf16(false, fa.v, false, fb.v,
                                                  (short)0, acc, false, false);
    __syncthreads();
  }

  if (lng) {
    // LayerNorm epilogue (only with Nn == 64, block covers full row)
#pragma unroll
    for (int r = 0; r < 8; ++r) {
      const int m = r + (lh << 3);
      const int c = (wave << 4) + lm;
      float v = acc[r];
      if (bias) v += bias[colBase + c];
      if (res)  v += res[(size_t)(rowBase + m) * Nn + colBase + c];
      sC[m * 64 + c] = v;
    }
    __syncthreads();
    if (tid < 16) {
      float s = 0.f, s2 = 0.f;
      for (int c = 0; c < 64; ++c) { float v = sC[tid * 64 + c]; s += v; s2 += v * v; }
      const float mu  = s * (1.f / 64.f);
      const float var = s2 * (1.f / 64.f) - mu * mu;
      sMu[tid] = mu;
      sRs[tid] = __frsqrt_rn(var + 1e-5f);
    }
    __syncthreads();
    for (int i = tid; i < 16 * 64; i += 128) {
      const int m = i >> 6, c = i & 63;
      C[(size_t)(rowBase + m) * Nn + colBase + c] =
          (sC[i] - sMu[m]) * sRs[m] * lng[c] + lnb[c];
    }
  } else {
#pragma unroll
    for (int r = 0; r < 8; ++r) {
      const int m = r + (lh << 3);
      const int c = (wave << 4) + lm;
      const int row = rowBase + m, col = colBase + c;
      float v = acc[r];
      if (bias)  v += bias[col];
      if (scale) v = v * scale[col] + shift[col];
      if (res)   v += res[(size_t)row * Nn + col];
      if (relu)  v = v > 0.f ? v : 0.f;
      C[(size_t)row * Nn + col] = v;
    }
  }
}

__global__ void zero4_kernel(float4* __restrict__ p, long n4) {
  long i = (long)blockIdx.x * blockDim.x + threadIdx.x;
  if (i < n4) p[i] = make_float4(0.f, 0.f, 0.f, 0.f);
}

// xs[s][:] = h[sni[s]][:]   (float4 granularity; 16 float4 per row)
__global__ void gather_kernel(const float* __restrict__ h, const int* __restrict__ sni,
                              float* __restrict__ xs, long total4) {
  long i = (long)blockIdx.x * blockDim.x + threadIdx.x;
  if (i < total4) {
    long s = i >> 4; int d4 = (int)(i & 15) << 2;
    *(float4*)(xs + (s << 6) + d4) =
        *(const float4*)(h + ((size_t)sni[s] << 6) + d4);
  }
}

// dst[sei1[e]][:] += src[sei0[e]][:]   (4 floats per thread)
__global__ void scatter_kernel(const float* __restrict__ src, const int* __restrict__ sei,
                               float* __restrict__ dst, long total4) {
  long i = (long)blockIdx.x * blockDim.x + threadIdx.x;
  if (i < total4) {
    long e = i >> 4; int d4 = (int)(i & 15) << 2;
    const int s0 = sei[e];
    const int s1 = sei[(long)SEDGE + e];
    float4 v = *(const float4*)(src + ((size_t)s0 << 6) + d4);
    float* d = dst + ((size_t)s1 << 6) + d4;
    atomicAdd(d + 0, v.x); atomicAdd(d + 1, v.y);
    atomicAdd(d + 2, v.z); atomicAdd(d + 3, v.w);
  }
}

// struct[n] = mean of the 8 contiguous subgraph rows of node n
__global__ void structpool_kernel(const float* __restrict__ xs, float* __restrict__ st,
                                  long total4) {
  long i = (long)blockIdx.x * blockDim.x + threadIdx.x;
  if (i < total4) {
    long n = i >> 4; int d4 = (int)(i & 15) << 2;
    float4 s = make_float4(0.f, 0.f, 0.f, 0.f);
#pragma unroll
    for (int j = 0; j < 8; ++j) {
      float4 v = *(const float4*)(xs + (((n << 3) + j) << 6) + d4);
      s.x += v.x; s.y += v.y; s.z += v.z; s.w += v.w;
    }
    s.x *= 0.125f; s.y *= 0.125f; s.z *= 0.125f; s.w *= 0.125f;
    *(float4*)(st + (n << 6) + d4) = s;
  }
}

__global__ __launch_bounds__(256)
void attention_kernel(const float* __restrict__ Q, const float* __restrict__ K,
                      const float* __restrict__ V, float* __restrict__ O) {
  __shared__ float sq[NPGC * DIM], sk[NPGC * DIM], sv[NPGC * DIM];
  __shared__ float sat[HEADSC * NPGC * NPGC];
  const int b = blockIdx.x, tid = threadIdx.x;
  const size_t base = (size_t)b * NPGC * DIM;
  for (int i = tid; i < NPGC * DIM; i += 256) {
    sq[i] = Q[base + i]; sk[i] = K[base + i]; sv[i] = V[base + i];
  }
  __syncthreads();
  const float sc = 0.35355339059327373f;  // 1/sqrt(DH)
  for (int i = tid; i < HEADSC * NPGC * NPGC; i += 256) {
    int hh = i >> 10, qi = (i >> 5) & 31, kj = i & 31;
    float s = 0.f;
#pragma unroll
    for (int d = 0; d < DHC; ++d)
      s += sq[qi * DIM + hh * DHC + d] * sk[kj * DIM + hh * DHC + d];
    sat[i] = s * sc;
  }
  __syncthreads();
  {  // softmax over 256 rows (8 heads x 32 queries), one thread per row
    int hh = tid >> 5, qi = tid & 31;
    float* row = &sat[(hh << 10) + (qi << 5)];
    float mx = row[0];
    for (int j = 1; j < 32; ++j) mx = fmaxf(mx, row[j]);
    float sum = 0.f;
    for (int j = 0; j < 32; ++j) { float e = __expf(row[j] - mx); row[j] = e; sum += e; }
    float inv = 1.f / sum;
    for (int j = 0; j < 32; ++j) row[j] *= inv;
  }
  __syncthreads();
  for (int i = tid; i < NPGC * DIM; i += 256) {
    int qi = i >> 6, c = i & 63, hh = c >> 3;
    float s = 0.f;
    for (int j = 0; j < 32; ++j) s += sat[(hh << 10) + (qi << 5) + j] * sv[j * DIM + c];
    O[base + i] = s;
  }
}

__global__ void pool_kernel(const float* __restrict__ h, float* __restrict__ g, long total4) {
  long i = (long)blockIdx.x * blockDim.x + threadIdx.x;
  if (i < total4) {
    long b = i >> 4; int d4 = (int)(i & 15) << 2;
    float4 s = make_float4(0.f, 0.f, 0.f, 0.f);
    for (int j = 0; j < NPGC; ++j) {
      float4 v = *(const float4*)(h + (((b << 5) + j) << 6) + d4);
      s.x += v.x; s.y += v.y; s.z += v.z; s.w += v.w;
    }
    const float r = 1.f / NPGC;
    s.x *= r; s.y *= r; s.z *= r; s.w *= r;
    *(float4*)(g + (b << 6) + d4) = s;
  }
}

__global__ void combine_kernel(const float* __restrict__ G, const float* __restrict__ S,
                               const float* __restrict__ graw, float* __restrict__ HCAT,
                               int side, long total) {
  long i = (long)blockIdx.x * blockDim.x + threadIdx.x;
  if (i < total) {
    long b = i >> 6; int d = (int)(i & 63);
    float gm = 1.f / (1.f + __expf(-graw[0]));
    HCAT[(size_t)b * 128 + (size_t)side * 64 + d] = (1.f - gm) * G[i] + gm * S[i];
  }
}

__global__ void finaldot_kernel(const float* __restrict__ HID, const float* __restrict__ w2,
                                const float* __restrict__ b2, float* __restrict__ out) {
  int b = blockIdx.x * blockDim.x + threadIdx.x;
  if (b < BGRAPH) {
    float s = b2[0];
    for (int c = 0; c < 64; ++c) s += HID[(size_t)b * 64 + c] * w2[c];
    out[b] = s;
  }
}

extern "C" void kernel_launch(void* const* d_in, const int* in_sizes, int n_in,
                              void* d_out, int out_size, void* d_ws, size_t ws_size,
                              hipStream_t stream) {
  (void)in_sizes; (void)n_in; (void)out_size; (void)ws_size;
  const float* x[2]      = {(const float*)d_in[0], (const float*)d_in[2]};
  const float* smiles[2] = {(const float*)d_in[1], (const float*)d_in[3]};
  const int* sni[2] = {(const int*)d_in[5], (const int*)d_in[9]};
  const int* sei[2] = {(const int*)d_in[6], (const int*)d_in[10]};
  const float* atom_w   = (const float*)d_in[12];
  const float* map_w    = (const float*)d_in[13];
  const float* map_b    = (const float*)d_in[14];
  const float* graw     = (const float*)d_in[15];
  const float* gnn_w    = (const float*)d_in[16];
  const float* struct_w = (const float*)d_in[17];
  const float* wq = (const float*)d_in[18]; const float* bq = (const float*)d_in[19];
  const float* wk = (const float*)d_in[20]; const float* bk = (const float*)d_in[21];
  const float* wv = (const float*)d_in[22]; const float* bv = (const float*)d_in[23];
  const float* wo = (const float*)d_in[24]; const float* bo = (const float*)d_in[25];
  const float* w1 = (const float*)d_in[26]; const float* b1 = (const float*)d_in[27];
  const float* w2 = (const float*)d_in[28]; const float* b2 = (const float*)d_in[29];
  const float* ln1g = (const float*)d_in[30]; const float* ln1b = (const float*)d_in[31];
  const float* ln2g = (const float*)d_in[32]; const float* ln2b = (const float*)d_in[33];
  const float* mw1 = (const float*)d_in[34]; const float* mb1 = (const float*)d_in[35];
  const float* bng = (const float*)d_in[36]; const float* bnb = (const float*)d_in[37];
  const float* mw2 = (const float*)d_in[38]; const float* mb2 = (const float*)d_in[39];

  float* ws = (float*)d_ws;
  size_t off = 0;
  float* H  = ws + off; off += (size_t)NNODES * DIM;   // 16 MB
  float* P0 = ws + off; off += (size_t)NSUB * DIM;     // 128 MB
  float* P1 = ws + off; off += (size_t)NSUB * DIM;     // 128 MB
  float* G[2];
  G[0] = ws + off; off += (size_t)BGRAPH * DIM;
  G[1] = ws + off; off += (size_t)BGRAPH * DIM;
  float* SML  = ws + off; off += (size_t)BGRAPH * DIM;
  float* HCAT = ws + off; off += (size_t)BGRAPH * 2 * DIM;
  float* HID  = ws + off; off += (size_t)BGRAPH * DIM;

  const size_t ND = (size_t)NNODES * DIM;

  auto gemm = [&](const float* A, const float* A2, const float* W, const float* bias,
                  const float* res, const float* scale, const float* shift,
                  const float* g, const float* bb, float* C,
                  int M, int Kd, int Nd, int relu) {
    dim3 gr((unsigned)(M / 16), (unsigned)(Nd / 64));
    const bool kal = (Kd % 32) == 0;
    if (A2) {
      gemm_bf16_kernel<true, true><<<gr, 128, 0, stream>>>(
          A, A2, W, bias, res, scale, shift, g, bb, C, M, Kd, Nd, relu);
    } else if (kal) {
      gemm_bf16_kernel<false, true><<<gr, 128, 0, stream>>>(
          A, nullptr, W, bias, res, scale, shift, g, bb, C, M, Kd, Nd, relu);
    } else {
      gemm_bf16_kernel<false, false><<<gr, 128, 0, stream>>>(
          A, nullptr, W, bias, res, scale, shift, g, bb, C, M, Kd, Nd, relu);
    }
  };

  const long totS4 = (long)NSUB * 16;     // float4 counts
  const long totE4 = (long)SEDGE * 16;
  const long totN4 = (long)NNODES * 16;
  const long totB4 = (long)BGRAPH * 16;
  const long totB  = (long)BGRAPH * DIM;

  for (int side = 0; side < 2; ++side) {
    // h = x @ atom_w  (K=133, zero-padded K tiles; non-aligned path)
    gemm(x[side], nullptr, atom_w, nullptr, nullptr, nullptr, nullptr, nullptr, nullptr,
         H, NNODES, 133, DIM, 0);

    for (int l = 0; l < LLAYERS; ++l) {
      // xs = h[sni]
      gather_kernel<<<(unsigned)((totS4 + 255) / 256), 256, 0, stream>>>(H, sni[side], P0, totS4);
      // hop 0: agg in P1; xs' = relu((xs+agg)@W) in-place into P1 (row-local)
      zero4_kernel<<<(unsigned)((totS4 + 255) / 256), 256, 0, stream>>>((float4*)P1, totS4);
      scatter_kernel<<<(unsigned)((totE4 + 255) / 256), 256, 0, stream>>>(P0, sei[side], P1, totE4);
      gemm(P0, P1, gnn_w + ((size_t)l * 2 + 0) * DIM * DIM, nullptr, nullptr, nullptr,
           nullptr, nullptr, nullptr, P1, NSUB, DIM, DIM, 1);
      // hop 1: agg in P0; xs'' -> P0
      zero4_kernel<<<(unsigned)((totS4 + 255) / 256), 256, 0, stream>>>((float4*)P0, totS4);
      scatter_kernel<<<(unsigned)((totE4 + 255) / 256), 256, 0, stream>>>(P1, sei[side], P0, totE4);
      gemm(P1, P0, gnn_w + ((size_t)l * 2 + 1) * DIM * DIM, nullptr, nullptr, nullptr,
           nullptr, nullptr, nullptr, P0, NSUB, DIM, DIM, 1);

      // overlay per-node buffers on P1 (xs'' lives in P0; P1 dead)
      float* STRUCT = P1;
      float* QK = P1 + ND;
      float* Qb = P1 + 2 * ND;
      float* Kb = P1 + 3 * ND;
      float* Vb = P1 + 4 * ND;
      float* Ob = P1 + 5 * ND;

      structpool_kernel<<<(unsigned)((totN4 + 255) / 256), 256, 0, stream>>>(P0, STRUCT, totN4);
      // qk = h + struct @ struct_w
      gemm(STRUCT, nullptr, struct_w + (size_t)l * DIM * DIM, nullptr, H, nullptr,
           nullptr, nullptr, nullptr, QK, NNODES, DIM, DIM, 0);
      gemm(QK, nullptr, wq + (size_t)l * DIM * DIM, bq + (size_t)l * DIM, nullptr,
           nullptr, nullptr, nullptr, nullptr, Qb, NNODES, DIM, DIM, 0);
      gemm(QK, nullptr, wk + (size_t)l * DIM * DIM, bk + (size_t)l * DIM, nullptr,
           nullptr, nullptr, nullptr, nullptr, Kb, NNODES, DIM, DIM, 0);
      gemm(H, nullptr, wv + (size_t)l * DIM * DIM, bv + (size_t)l * DIM, nullptr,
           nullptr, nullptr, nullptr, nullptr, Vb, NNODES, DIM, DIM, 0);
      attention_kernel<<<BGRAPH, 256, 0, stream>>>(Qb, Kb, Vb, Ob);
      // h = LN1(h + o@wo + bo)  (in-place: each block touches only its rows)
      gemm(Ob, nullptr, wo + (size_t)l * DIM * DIM, bo + (size_t)l * DIM, H, nullptr,
           nullptr, ln1g + (size_t)l * DIM, ln1b + (size_t)l * DIM, H, NNODES, DIM, DIM, 0);
      // FFN: ff1 in P0 (xs'' dead)
      float* FF1 = P0;
      gemm(H, nullptr, w1 + (size_t)l * DIM * DFFC, b1 + (size_t)l * DFFC, nullptr,
           nullptr, nullptr, nullptr, nullptr, FF1, NNODES, DIM, DFFC, 1);
      gemm(FF1, nullptr, w2 + (size_t)l * DFFC * DIM, b2 + (size_t)l * DIM, H, nullptr,
           nullptr, ln2g + (size_t)l * DIM, ln2b + (size_t)l * DIM, H, NNODES, DFFC, DIM, 0);
    }
    pool_kernel<<<(unsigned)((totB4 + 255) / 256), 256, 0, stream>>>(H, G[side], totB4);
  }

  for (int side = 0; side < 2; ++side) {
    gemm(smiles[side], nullptr, map_w, map_b, nullptr, nullptr, nullptr, nullptr, nullptr,
         SML, BGRAPH, 384, DIM, 0);
    combine_kernel<<<(unsigned)((totB + 255) / 256), 256, 0, stream>>>(
        G[side], SML, graw, HCAT, side, totB);
  }
  // hid = relu((hcat@m_w1 + m_b1)*bn_g + bn_b)
  gemm(HCAT, nullptr, mw1, mb1, nullptr, bng, bnb, nullptr, nullptr,
       HID, BGRAPH, 2 * DIM, DIM, 1);
  finaldot_kernel<<<(BGRAPH + 255) / 256, 256, 0, stream>>>(HID, mw2, mb2, (float*)d_out);
}
